// SelfMultiheadAttention_35107062677896
// MI455X (gfx1250) — matrix operand order repaired and checked by reading, not compiled, not run on hardware
//
#include <hip/hip_runtime.h>
#include <hip/hip_bf16.h>

// ---------------------------------------------------------------------------
// Self multi-head attention, MI455X (gfx1250, wave32, WMMA bf16)
// B=2, S=2048, E=1024, H=16, D=64
// ---------------------------------------------------------------------------

#if defined(__has_builtin)
#if __has_builtin(__builtin_amdgcn_global_load_async_to_lds_b128)
#define HAVE_ASYNC_LDS 1
#endif
#endif
#ifndef HAVE_ASYNC_LDS
#define HAVE_ASYNC_LDS 0
#endif

namespace {

constexpr int Bn = 2;
constexpr int S  = 2048;
constexpr int E  = 1024;
constexpr int H  = 16;
constexpr int D  = 64;
constexpr float NEG_INF = -1e9f;

typedef __attribute__((ext_vector_type(16))) __bf16 v16bf;
typedef __attribute__((ext_vector_type(8)))  __bf16 v8bf;
typedef __attribute__((ext_vector_type(8)))  float  v8f;
typedef __attribute__((ext_vector_type(4)))  int    v4i;

__device__ __forceinline__ v8f wmma_bf16(v16bf a, v16bf b, v8f c) {
  // (neg_a, A, neg_b, B, c_mod, C, reuse_a, reuse_b)
  return __builtin_amdgcn_wmma_f32_16x16x32_bf16(false, a, false, b,
                                                 (short)0, c, false, false);
}

__device__ __forceinline__ v16bf concat8(v8bf lo, v8bf hi) {
  return __builtin_shufflevector(lo, hi, 0, 1, 2, 3, 4, 5, 6, 7, 8, 9, 10, 11,
                                 12, 13, 14, 15);
}

// A-matrix 16x32 bf16 pack (ISA 7.12.2): lane holds row m = lane%16,
// K = 8*half + [0..7] then 16 + 8*half + [0..7].  Two 16B vector loads.
__device__ __forceinline__ v16bf pack_a(const __bf16* __restrict__ base,
                                        int ld, int m, int k0, int half) {
  const __bf16* r = base + (size_t)m * ld + k0 + 8 * half;
  return concat8(*(const v8bf*)r, *(const v8bf*)(r + 16));
}

// B-matrix 32x16 bf16 pack: lane holds col n = lane%16,
// K = 8*half + [0..7] then 16 + 8*half + [0..7].  Two 16B vector loads.
__device__ __forceinline__ v16bf pack_b(const __bf16* __restrict__ base,
                                        int ld, int n, int k0, int half) {
  const __bf16* r = base + (size_t)n * ld + k0 + 8 * half;
  return concat8(*(const v8bf*)r, *(const v8bf*)(r + 16));
}

// 16-byte global -> LDS copy; async (ASYNCcnt-tracked) when available.
// Builtin signature (probe-verified): (int4 AS1*, int4 AS3*, imm off, imm cpol)
__device__ __forceinline__ void copy16_g2l(const __bf16* g, __bf16* l) {
#if HAVE_ASYNC_LDS
  typedef __attribute__((address_space(1))) v4i* gp_t;
  typedef __attribute__((address_space(3))) v4i* lp_t;
  __builtin_amdgcn_global_load_async_to_lds_b128((gp_t)g, (lp_t)l, 0, 0);
#else
  *(uint4*)l = *(const uint4*)g;
#endif
}

template <int N>
__device__ __forceinline__ void wait_async() {
#if HAVE_ASYNC_LDS
#if __has_builtin(__builtin_amdgcn_s_wait_asynccnt)
  __builtin_amdgcn_s_wait_asynccnt((unsigned short)N);
#else
  asm volatile("s_wait_asynccnt %0" ::"i"(N) : "memory");
#endif
#endif
}

// ---------------------------------------------------------------------------
// Kernel 0: elementwise f32 -> bf16 (8 elements / thread, exact grid)
// ---------------------------------------------------------------------------
__global__ __launch_bounds__(256) void cvt_bf16_kernel(
    const float* __restrict__ src, __bf16* __restrict__ dst, int n) {
  const int i = (blockIdx.x * 256 + threadIdx.x) * 8;
  if (i >= n) return;
  const float4 a = *(const float4*)(src + i);
  const float4 b = *(const float4*)(src + i + 4);
  v8bf o;
  o[0] = (__bf16)a.x; o[1] = (__bf16)a.y; o[2] = (__bf16)a.z; o[3] = (__bf16)a.w;
  o[4] = (__bf16)b.x; o[5] = (__bf16)b.y; o[6] = (__bf16)b.z; o[7] = (__bf16)b.w;
  *(v8bf*)(dst + i) = o;
}

// ---------------------------------------------------------------------------
// Kernel 1: fused QKV projection (bf16 GEMM).  qkv = query @ W_in^T + b_in
// Register-double-buffered k-loop: next step's A + 4 B packs are prefetched
// before the current 4 WMMAs, so the loads overlap matrix work and the wait
// before the WMMA group leaves the in-flight prefetches outstanding.
// Writes Q (scaled by D^-0.5) and K as bf16 [B,H,S,D]; V transposed [B,H,D,S].
// ---------------------------------------------------------------------------
__global__ __launch_bounds__(128) void qkv_proj_kernel(
    const __bf16* __restrict__ qbf, const __bf16* __restrict__ wbf,
    const float* __restrict__ bias, __bf16* __restrict__ Qs,
    __bf16* __restrict__ Ks, __bf16* __restrict__ Vt) {
  const int lane = threadIdx.x & 31;
  const int half = lane >> 4;
  const int ln   = lane & 15;
  const int wid  = threadIdx.x >> 5;
  const int wg   = blockIdx.x * 4 + wid;          // 12288 waves
  const int row_tile = wg / 48;                   // B*S/16 = 256
  const int colgrp   = wg % 48;                   // 48 * 64 = 3E
  const int m0 = row_tile * 16;
  const int n0 = colgrp * 64;
  const int m  = m0 + ln;

  v8f acc[4] = {v8f{}, v8f{}, v8f{}, v8f{}};
  v16bf a_c = pack_a(qbf, E, m, 0, half);
  v16bf b_c[4];
#pragma unroll
  for (int t = 0; t < 4; ++t) b_c[t] = pack_b(wbf, E, n0 + t * 16 + ln, 0, half);

#pragma unroll 2
  for (int k0 = 0; k0 < E - 32; k0 += 32) {
    // prefetch next k-step into the shadow register set
    const v16bf a_n = pack_a(qbf, E, m, k0 + 32, half);
    v16bf b_n[4];
#pragma unroll
    for (int t = 0; t < 4; ++t)
      b_n[t] = pack_b(wbf, E, n0 + t * 16 + ln, k0 + 32, half);
    // consume current registers while prefetch loads are in flight
#pragma unroll
    for (int t = 0; t < 4; ++t) acc[t] = wmma_bf16(a_c, b_c[t], acc[t]);
    a_c = a_n;
#pragma unroll
    for (int t = 0; t < 4; ++t) b_c[t] = b_n[t];
  }
#pragma unroll
  for (int t = 0; t < 4; ++t) acc[t] = wmma_bf16(a_c, b_c[t], acc[t]);

#pragma unroll
  for (int t = 0; t < 4; ++t) {
    const int c   = n0 + t * 16 + ln;
    const float bv = bias[c];
    const int qkv = c >> 10;          // 0=Q 1=K 2=V
    const int e   = c & 1023;
    const int h   = e >> 6;
    const int d   = e & 63;
#pragma unroll
    for (int r = 0; r < 8; ++r) {
      const int row = m0 + r + 8 * half;  // C layout: M = r + 8*half
      const int b   = row >> 11;
      const int s   = row & 2047;
      const float v = acc[t][r] + bv;
      const size_t bh = (size_t)(b * H + h);
      if (qkv == 0)      Qs[(bh * S + s) * D + d] = (__bf16)(v * 0.125f);
      else if (qkv == 1) Ks[(bh * S + s) * D + d] = (__bf16)v;
      else               Vt[(bh * D + d) * S + s] = (__bf16)v;
    }
  }
}

// ---------------------------------------------------------------------------
// Kernel 2: flash attention.  One block (4 waves) owns 64 queries of one
// (b,h); waves hold 16-query strips.  Per 32-key step the block stages the
// K tile [32x64] and V^T tile [64x32] into double-buffered LDS with async
// global->LDS copies (ASYNCcnt + barrier), then each wave runs 4 WMMA for
// scores, mask/bias (bias streamed NT: the 512 MiB roofline term), online
// softmax (shfl_xor within 16-lane halves), P restaged via LDS, 4 WMMA P@V.
// ---------------------------------------------------------------------------
__global__ __launch_bounds__(128) void attn_kernel(
    const __bf16* __restrict__ Qs, const __bf16* __restrict__ Ks,
    const __bf16* __restrict__ Vt, const int* __restrict__ kpm,
    const float* __restrict__ attn_bias, __bf16* __restrict__ Obuf) {
  __shared__ __bf16 Ktile[2][32 * 64];  // [buf][key][d]   8 KiB
  __shared__ __bf16 Vtile[2][64 * 32];  // [buf][d][key]   8 KiB
  __shared__ __bf16 Plds[4][16 * 32];   // per-wave P      4 KiB

  const int tid  = threadIdx.x;
  const int lane = tid & 31;
  const int half = lane >> 4;
  const int ln   = lane & 15;
  const int wid  = tid >> 5;
  const int bh   = blockIdx.x >> 5;   // B*H = 32
  const int qc   = blockIdx.x & 31;   // S/64 = 32 query chunks
  const int b    = bh >> 4;
  const int h    = bh & 15;
  const int q0   = qc * 64 + wid * 16;

  const __bf16* Qbase = Qs + (size_t)bh * S * D;
  const __bf16* Kbase = Ks + (size_t)bh * S * D;
  const __bf16* Vbase = Vt + (size_t)bh * D * S;
  const float* bias_base = attn_bias + (size_t)bh * S * S;
  const int* mrow = kpm + b * S;

  // cooperative stage of one K/V tile pair: 512 x 16B chunks over 128 threads
  auto stage = [&](int buf, int key0) {
    __bf16* kt = &Ktile[buf][0];
    __bf16* vt = &Vtile[buf][0];
#pragma unroll
    for (int j = 0; j < 2; ++j) {
      const int c  = tid + 128 * j;
      const int kr = c >> 3, ko = (c & 7) * 8;   // K: 8 chunks per 64-el row
      copy16_g2l(Kbase + (size_t)(key0 + kr) * D + ko, kt + c * 8);
      const int vr = c >> 2, vo = (c & 3) * 8;   // V: 4 chunks per 32-el row
      copy16_g2l(Vbase + (size_t)vr * S + key0 + vo, vt + c * 8);
    }
  };

  // Q strip loaded once (already scaled by D^-0.5)
  v16bf qa[2];
  qa[0] = pack_a(Qbase, D, q0 + ln, 0, half);
  qa[1] = pack_a(Qbase, D, q0 + ln, 32, half);

  float mstate[8], lstate[8];
  v8f o[4] = {v8f{}, v8f{}, v8f{}, v8f{}};
#pragma unroll
  for (int r = 0; r < 8; ++r) { mstate[r] = -__builtin_inff(); lstate[r] = 0.f; }

  __bf16* pl = &Plds[wid][0];
  constexpr int NT = S / 32;  // 64 key tiles

  stage(0, 0);  // prologue

  for (int i = 0; i < NT; ++i) {
    const int key0 = i * 32;
    const int buf  = i & 1;
    if (i + 1 < NT) {
      stage(buf ^ 1, key0 + 32);  // 4 async ops/thread into the other buffer
      wait_async<4>();            // previous tile's copies have landed
    } else {
      wait_async<0>();
    }
    __syncthreads();              // tile `buf` visible to all waves

    // ---- scores: 16x32 = Q(16xD) @ K^T (K-dim = d, two 32-chunks) ----
    const __bf16* kt = &Ktile[buf][0];
    v8f c0 = v8f{}, c1 = v8f{};
#pragma unroll
    for (int p = 0; p < 2; ++p) {
      const v16bf kb0 = pack_b(kt, D, ln, 32 * p, half);
      const v16bf kb1 = pack_b(kt, D, 16 + ln, 32 * p, half);
      c0 = wmma_bf16(qa[p], kb0, c0);
      c1 = wmma_bf16(qa[p], kb1, c1);
    }

    // ---- key padding mask then additive bias (reference order) ----
    const int key_a = key0 + ln;
    const int key_b = key0 + 16 + ln;
    const bool mk0 = mrow[key_a] != 0;
    const bool mk1 = mrow[key_b] != 0;
    float s0[8], s1[8];
#pragma unroll
    for (int r = 0; r < 8; ++r) {
      const int srow = q0 + r + 8 * half;
      const float b0 =
          __builtin_nontemporal_load(bias_base + (size_t)srow * S + key_a);
      const float b1 =
          __builtin_nontemporal_load(bias_base + (size_t)srow * S + key_b);
      s0[r] = (mk0 ? NEG_INF : c0[r]) + b0;
      s1[r] = (mk1 ? NEG_INF : c1[r]) + b1;
    }
    if (key0 + 32 < S)  // stream-ahead for the dominant HBM term
      __builtin_prefetch(bias_base + (size_t)(q0 + 8 * half) * S + key0 + 32,
                         0, 0);

    // ---- online softmax: each row lives within one 16-lane half ----
    float alpha[8];
#pragma unroll
    for (int r = 0; r < 8; ++r) {
      float v = fmaxf(s0[r], s1[r]);
      v = fmaxf(v, __shfl_xor(v, 1, 32));
      v = fmaxf(v, __shfl_xor(v, 2, 32));
      v = fmaxf(v, __shfl_xor(v, 4, 32));
      v = fmaxf(v, __shfl_xor(v, 8, 32));
      const float mnew = fmaxf(mstate[r], v);
      alpha[r] = __expf(mstate[r] - mnew);
      const float p0 = __expf(s0[r] - mnew);
      const float p1 = __expf(s1[r] - mnew);
      float rs = p0 + p1;
      rs += __shfl_xor(rs, 1, 32);
      rs += __shfl_xor(rs, 2, 32);
      rs += __shfl_xor(rs, 4, 32);
      rs += __shfl_xor(rs, 8, 32);
      lstate[r] = lstate[r] * alpha[r] + rs;
      mstate[r] = mnew;
      s0[r] = p0;
      s1[r] = p1;
    }
#pragma unroll
    for (int t = 0; t < 4; ++t)
#pragma unroll
      for (int r = 0; r < 8; ++r) o[t][r] *= alpha[r];

    // ---- restage P (C-layout) into LDS as row-major 16x32 ----
#pragma unroll
    for (int r = 0; r < 8; ++r) {
      pl[(r + 8 * half) * 32 + ln]      = (__bf16)s0[r];
      pl[(r + 8 * half) * 32 + 16 + ln] = (__bf16)s1[r];
    }
    const v16bf pa = pack_a(pl, 32, ln, 0, half);  // K = 32 local keys

    // ---- O += P @ V from the staged V^T tile ----
    const __bf16* vt = &Vtile[buf][0];
#pragma unroll
    for (int dt = 0; dt < 4; ++dt) {
      const v16bf vb = pack_b(vt, 32, dt * 16 + ln, 0, half);
      o[dt] = wmma_bf16(pa, vb, o[dt]);
    }
    __syncthreads();  // all waves done reading buf before it is refilled
  }

  // ---- epilogue: O /= l, store bf16 [B,S,E] with E = h*64 + dt*16 + n ----
#pragma unroll
  for (int dt = 0; dt < 4; ++dt) {
    const int e = h * 64 + dt * 16 + ln;
#pragma unroll
    for (int r = 0; r < 8; ++r) {
      const int s = q0 + r + 8 * half;
      Obuf[((size_t)b * S + s) * E + e] = (__bf16)(o[dt][r] / lstate[r]);
    }
  }
}

// ---------------------------------------------------------------------------
// Kernel 3: output projection (bf16 GEMM, register-double-buffered k-loop).
// out = O @ W_out^T + b_out  (f32 output)
// ---------------------------------------------------------------------------
__global__ __launch_bounds__(128) void out_proj_kernel(
    const __bf16* __restrict__ Obuf, const __bf16* __restrict__ wbf,
    const float* __restrict__ bias, float* __restrict__ out) {
  const int lane = threadIdx.x & 31;
  const int half = lane >> 4;
  const int ln   = lane & 15;
  const int wid  = threadIdx.x >> 5;
  const int wg   = blockIdx.x * 4 + wid;  // 4096 waves
  const int row_tile = wg >> 4;           // 256
  const int colgrp   = wg & 15;           // 16 * 64 = E
  const int m0 = row_tile * 16;
  const int n0 = colgrp * 64;
  const int m  = m0 + ln;

  v8f acc[4] = {v8f{}, v8f{}, v8f{}, v8f{}};
  v16bf a_c = pack_a(Obuf, E, m, 0, half);
  v16bf b_c[4];
#pragma unroll
  for (int t = 0; t < 4; ++t) b_c[t] = pack_b(wbf, E, n0 + t * 16 + ln, 0, half);

#pragma unroll 2
  for (int k0 = 0; k0 < E - 32; k0 += 32) {
    const v16bf a_n = pack_a(Obuf, E, m, k0 + 32, half);
    v16bf b_n[4];
#pragma unroll
    for (int t = 0; t < 4; ++t)
      b_n[t] = pack_b(wbf, E, n0 + t * 16 + ln, k0 + 32, half);
#pragma unroll
    for (int t = 0; t < 4; ++t) acc[t] = wmma_bf16(a_c, b_c[t], acc[t]);
    a_c = a_n;
#pragma unroll
    for (int t = 0; t < 4; ++t) b_c[t] = b_n[t];
  }
#pragma unroll
  for (int t = 0; t < 4; ++t) acc[t] = wmma_bf16(a_c, b_c[t], acc[t]);

#pragma unroll
  for (int t = 0; t < 4; ++t) {
    const int c   = n0 + t * 16 + ln;
    const float bv = bias[c];
#pragma unroll
    for (int r = 0; r < 8; ++r) {
      const int row = m0 + r + 8 * half;
      out[(size_t)row * E + c] = acc[t][r] + bv;
    }
  }
}

}  // namespace

extern "C" void kernel_launch(void* const* d_in, const int* in_sizes, int n_in,
                              void* d_out, int out_size, void* d_ws,
                              size_t ws_size, hipStream_t stream) {
  const float* query     = (const float*)d_in[0];
  const int*   kpm       = (const int*)d_in[1];
  const float* attn_bias = (const float*)d_in[2];
  const float* in_w      = (const float*)d_in[3];
  const float* in_b      = (const float*)d_in[4];
  const float* out_w     = (const float*)d_in[5];
  const float* out_b     = (const float*)d_in[6];
  float* out = (float*)d_out;

  // Scratch layout (48 MiB total):
  //   qbf   8 MiB | winbf 6 MiB | woutbf 2 MiB | Qs 8 | Ks 8 | Vt 8 | Ob 8
  char* ws = (char*)d_ws;
  const size_t MB = 1024 * 1024;
  __bf16* qbf    = (__bf16*)(ws + 0 * MB);
  __bf16* winbf  = (__bf16*)(ws + 8 * MB);
  __bf16* woutbf = (__bf16*)(ws + 14 * MB);
  __bf16* Qs     = (__bf16*)(ws + 16 * MB);
  __bf16* Ks     = (__bf16*)(ws + 24 * MB);
  __bf16* Vt     = (__bf16*)(ws + 32 * MB);
  __bf16* Ob     = (__bf16*)(ws + 40 * MB);

  const int n_q  = Bn * S * E;      // 4,194,304
  const int n_wi = 3 * E * E;       // 3,145,728
  const int n_wo = E * E;           // 1,048,576

  // 0) one-time f32 -> bf16 conversions (removes cvt chains from GEMM loops)
  cvt_bf16_kernel<<<n_q / (8 * 256), 256, 0, stream>>>(query, qbf, n_q);
  cvt_bf16_kernel<<<n_wi / (8 * 256), 256, 0, stream>>>(in_w, winbf, n_wi);
  cvt_bf16_kernel<<<n_wo / (8 * 256), 256, 0, stream>>>(out_w, woutbf, n_wo);
  // 1) fused QKV projection: (B*S/16)*(3E/64) = 12288 waves
  qkv_proj_kernel<<<3072, 128, 0, stream>>>(qbf, winbf, in_b, Qs, Ks, Vt);
  // 2) flash attention: B*H*(S/64) blocks, 4 waves each
  attn_kernel<<<1024, 128, 0, stream>>>(Qs, Ks, Vt, kpm, attn_bias, Ob);
  // 3) output projection: (B*S/16)*(E/64) = 4096 waves
  out_proj_kernel<<<1024, 128, 0, stream>>>(Ob, woutbf, out_b, out);
}
